// HybridMoEBlock_11330123727004
// MI455X (gfx1250) — matrix operand order repaired and checked by reading, not compiled, not run on hardware
//
#include <hip/hip_runtime.h>

typedef __attribute__((ext_vector_type(16))) __bf16 bf16x16;
typedef __attribute__((ext_vector_type(2)))  __bf16 bf16x2;
typedef __attribute__((ext_vector_type(8)))  float  f32x8;

#define DEV __device__ __forceinline__

constexpr int T_TOK = 8192;   // B*S
constexpr int DIM   = 1024;   // D
constexpr int FFD   = 4096;   // FF
constexpr int NEXP  = 8;      // E
constexpr int CHUNK = 1024;   // FF processed per pass (bounds H buffer)
constexpr int BM = 128, BN = 128, BK = 32;
constexpr int LDSB = 40;      // padded row stride (ushorts): 80B, 16B-aligned, conflict-free
constexpr int SLOTS = 2 * T_TOK + NEXP * BM;   // 17408: 2T pairs + per-expert pad

// ---- workspace layout (bytes) ----
constexpr size_t O_RP   = 0;                                  // rp      [T,2] f32
constexpr size_t O_TE   = O_RP  + (size_t)T_TOK * 2 * 4;      // tok_e   [T,2] i32
constexpr size_t O_TW   = O_TE  + (size_t)T_TOK * 2 * 4;      // tok_w   [T,2] f32
constexpr size_t O_TS   = O_TW  + (size_t)T_TOK * 2 * 4;      // tok2slot[T,2] i32
constexpr size_t O_CNT  = O_TS  + (size_t)T_TOK * 2 * 4;      // counts  [E]
constexpr size_t O_OFF  = O_CNT + 256;                        // offsets [E+1]
constexpr size_t O_CUR  = O_OFF + 256;                        // cursors [E]
constexpr size_t O_LIST = O_CUR + 256;                        // slot->token [SLOTS]
constexpr size_t O_H    = O_LIST + (size_t)SLOTS * 4;         // H [SLOTS,CHUNK] bf16
constexpr size_t O_PO   = O_H   + (size_t)SLOTS * CHUNK * 2;  // pair_out [SLOTS,D] f32
constexpr size_t O_DA   = O_PO  + (size_t)SLOTS * DIM * 4;    // dense_acc [T,D] f32

DEV unsigned pk2(float a, float b) {          // packed fp32->bf16 pair (native cvt)
  bf16x2 t;
  t[0] = (__bf16)a;
  t[1] = (__bf16)b;
  return __builtin_bit_cast(unsigned, t);
}
DEV unsigned short bf1(float a) {             // single fp32->bf16 (native cvt)
  return __builtin_bit_cast(unsigned short, (__bf16)a);
}

union FragU { bf16x16 v; uint4 q[2]; };

// ---------------------------------------------------------------------------
// Router (softmax over 2) + MoE gate (softmax over 8, top-2, renormalize).
// One wave32 per token; lane-parallel dot products over D with xor-reduce.
// ---------------------------------------------------------------------------
__global__ __launch_bounds__(256) void router_gate(
    const float* __restrict__ x, const float* __restrict__ Wr,
    const float* __restrict__ br, const float* __restrict__ Wg,
    const float* __restrict__ bg, float* __restrict__ rp,
    int* __restrict__ tok_e, float* __restrict__ tok_w, int* __restrict__ counts)
{
  const int t    = blockIdx.x * 8 + (threadIdx.x >> 5);
  const int lane = threadIdx.x & 31;
  if (t >= T_TOK) return;

  float r0 = 0.f, r1 = 0.f, g[NEXP];
  #pragma unroll
  for (int e = 0; e < NEXP; ++e) g[e] = 0.f;

  const float* xr = x + (long long)t * DIM;
  for (int k = lane; k < DIM; k += 32) {
    float xv = xr[k];
    r0 += xv * Wr[k * 2 + 0];
    r1 += xv * Wr[k * 2 + 1];
    #pragma unroll
    for (int e = 0; e < NEXP; ++e) g[e] += xv * Wg[k * NEXP + e];
  }
  #pragma unroll
  for (int off = 16; off > 0; off >>= 1) {
    r0 += __shfl_xor(r0, off);
    r1 += __shfl_xor(r1, off);
    #pragma unroll
    for (int e = 0; e < NEXP; ++e) g[e] += __shfl_xor(g[e], off);
  }
  if (lane == 0) {
    r0 += br[0]; r1 += br[1];
    float m  = fmaxf(r0, r1);
    float e0 = __expf(r0 - m), e1 = __expf(r1 - m);
    float inv = 1.f / (e0 + e1);
    rp[t * 2 + 0] = e0 * inv;
    rp[t * 2 + 1] = e1 * inv;

    #pragma unroll
    for (int e = 0; e < NEXP; ++e) g[e] += bg[e];
    int i0 = 0;
    #pragma unroll
    for (int e = 1; e < NEXP; ++e) if (g[e] > g[i0]) i0 = e;   // ties -> lowest idx
    int i1 = (i0 == 0) ? 1 : 0;
    #pragma unroll
    for (int e = 0; e < NEXP; ++e) if (e != i1 && e != i0 && g[e] > g[i1]) i1 = e;
    // renormalized top-2 softmax weights == softmax over {g[i0], g[i1]}
    float p1 = __expf(g[i1] - g[i0]);
    float w0 = 1.f / (1.f + p1);
    tok_e[t * 2 + 0] = i0;  tok_e[t * 2 + 1] = i1;
    tok_w[t * 2 + 0] = w0;  tok_w[t * 2 + 1] = 1.f - w0;
    atomicAdd(&counts[i0], 1);
    atomicAdd(&counts[i1], 1);
  }
}

__global__ void init_small(int* counts, int* cursors, int* tlist) {
  int i = blockIdx.x * 256 + threadIdx.x;
  if (i < NEXP) { counts[i] = 0; cursors[i] = 0; }
  if (i < SLOTS) tlist[i] = 0;     // pad slots gather token 0 (outputs unread)
}

__global__ void make_offsets(const int* counts, int* offsets) {
  if (threadIdx.x == 0 && blockIdx.x == 0) {
    int acc = 0;
    for (int e = 0; e < NEXP; ++e) {
      offsets[e] = acc;
      acc += ((counts[e] + BM - 1) / BM) * BM;   // pad each expert to BM rows
    }
    offsets[NEXP] = acc;
  }
}

__global__ void scatter_tokens(const int* __restrict__ tok_e,
                               const int* __restrict__ offsets,
                               int* __restrict__ cursors,
                               int* __restrict__ tlist,
                               int* __restrict__ tok2slot)
{
  int t = blockIdx.x * 256 + threadIdx.x;
  if (t >= T_TOK) return;
  #pragma unroll
  for (int k = 0; k < 2; ++k) {
    int e    = tok_e[t * 2 + k];
    int slot = offsets[e] + atomicAdd(&cursors[e], 1);
    tlist[slot] = t;
    tok2slot[t * 2 + k] = slot;
  }
}

__global__ void init_pairout(float* __restrict__ P, const float* __restrict__ b2,
                             const int* __restrict__ offsets)
{
  size_t idx = (size_t)blockIdx.x * 256 + threadIdx.x;
  if (idx >= (size_t)SLOTS * DIM) return;
  int s = (int)(idx >> 10), d = (int)(idx & (DIM - 1));
  float v = 0.f;
  if (s < offsets[NEXP]) {
    int e = 0;
    while (offsets[e + 1] <= s) ++e;
    v = b2[e * DIM + d];
  }
  P[idx] = v;
}

__global__ void init_dense(float* __restrict__ A, const float* __restrict__ d2b) {
  size_t idx = (size_t)blockIdx.x * 256 + threadIdx.x;
  if (idx >= (size_t)T_TOK * DIM) return;
  A[idx] = d2b[idx & (DIM - 1)];
}

// ---------------------------------------------------------------------------
// GEMM pass 1: H[rows, chunk] = relu( gather(X)[rows, D] * W[:, ff0:+CHUNK] + bias )
// bf16 WMMA 16x16x32, f32 acc. 256 thr = 8 waves (4 row-strips x 2 col-halves),
// 128x128 tile, double-buffered LDS, all fragments fed by ds_load_b128.
// ---------------------------------------------------------------------------
__global__ __launch_bounds__(256) void gemm_ffn1(
    const float* __restrict__ X, int lda,
    const float* __restrict__ W, long long wStride, int ldw,
    const float* __restrict__ bias, int biasStride,
    unsigned short* __restrict__ H, int ldh,
    const int* __restrict__ list, const int* __restrict__ offsets,
    int rowsTotal, int ff0)
{
  __shared__ __align__(16) unsigned short As[2][BM][LDSB];  // row-major A tile
  __shared__ __align__(16) unsigned short Bt[2][BN][LDSB];  // TRANSPOSED B tile

  const int m0 = blockIdx.x * BM;
  int e = 0;
  if (offsets) {
    if (m0 >= offsets[NEXP]) return;
    while (offsets[e + 1] <= m0) ++e;       // tiles never cross experts (BM-padded)
  } else if (m0 >= rowsTotal) return;

  const int ncol = blockIdx.y * BN;         // chunk-local output column base
  const int tid = threadIdx.x, lane = tid & 31, lr = lane & 15, hi = lane >> 4;
  const int wv = tid >> 5;
  const int rs = (wv >> 1) * 32, cs = (wv & 1) * 64;   // wave's 32x64 sub-tile

  // A loader: thread -> (row, 16-col half)
  const int arow = tid >> 1, acb = (tid & 1) * 16;
  const int grow = list ? list[m0 + arow] : (m0 + arow);
  const float* Ap = X + (long long)grow * lda + acb;
  // B loader: thread -> 4 K-rows x 4 N-cols sub-tile, stored transposed
  const int kq = tid >> 5, nb = (tid & 31) * 4;
  const float* Bp = W + (long long)e * wStride + (ff0 + ncol) + nb;

  float ra[16], rb[4][4];
  auto loadA = [&](int k0) {
    #pragma unroll
    for (int i = 0; i < 4; ++i) {
      float4 v = *(const float4*)(Ap + k0 + i * 4);
      ra[i*4+0] = v.x; ra[i*4+1] = v.y; ra[i*4+2] = v.z; ra[i*4+3] = v.w;
    }
  };
  auto loadB = [&](int k0) {
    #pragma unroll
    for (int i = 0; i < 4; ++i) {
      float4 v = *(const float4*)(Bp + (long long)(k0 + kq * 4 + i) * ldw);
      rb[i][0] = v.x; rb[i][1] = v.y; rb[i][2] = v.z; rb[i][3] = v.w;
    }
  };
  auto stage = [&](int b) {
    uint4 q0, q1;
    q0.x = pk2(ra[0],  ra[1]);  q0.y = pk2(ra[2],  ra[3]);
    q0.z = pk2(ra[4],  ra[5]);  q0.w = pk2(ra[6],  ra[7]);
    q1.x = pk2(ra[8],  ra[9]);  q1.y = pk2(ra[10], ra[11]);
    q1.z = pk2(ra[12], ra[13]); q1.w = pk2(ra[14], ra[15]);
    *(uint4*)&As[b][arow][acb]     = q0;          // ds_store_b128
    *(uint4*)&As[b][arow][acb + 8] = q1;
    #pragma unroll
    for (int j = 0; j < 4; ++j) {                 // transpose: Bt[n][k0..k3]
      uint2 w;
      w.x = pk2(rb[0][j], rb[1][j]);
      w.y = pk2(rb[2][j], rb[3][j]);
      *(uint2*)&Bt[b][nb + j][kq * 4] = w;        // ds_store_b64
    }
  };

  f32x8 acc[2][4];
  #pragma unroll
  for (int a = 0; a < 2; ++a)
    #pragma unroll
    for (int c = 0; c < 4; ++c)
      #pragma unroll
      for (int i = 0; i < 8; ++i) acc[a][c][i] = 0.f;

  loadA(0); loadB(0); stage(0);
  int buf = 0;
  for (int k0 = 0; k0 < DIM; k0 += BK) {
    const bool nxt = (k0 + BK) < DIM;
    if (nxt) { loadA(k0 + BK); loadB(k0 + BK); }          // overlap with compute
    if (k0 + 2 * BK < DIM) {
      __builtin_prefetch(Ap + k0 + 2 * BK, 0, 1);          // global_prefetch_b8
      __builtin_prefetch(Bp + (long long)(k0 + 2 * BK + kq * 4) * ldw, 0, 1);
    }
    __syncthreads();

    FragU a0, a1;                                          // 2x ds_load_b128 each
    a0.q[0] = *(const uint4*)&As[buf][rs + lr][8 * hi];
    a0.q[1] = *(const uint4*)&As[buf][rs + lr][16 + 8 * hi];
    a1.q[0] = *(const uint4*)&As[buf][rs + 16 + lr][8 * hi];
    a1.q[1] = *(const uint4*)&As[buf][rs + 16 + lr][16 + 8 * hi];
    #pragma unroll
    for (int bc = 0; bc < 4; ++bc) {
      FragU bfr;
      const int col = cs + bc * 16 + lr;
      bfr.q[0] = *(const uint4*)&Bt[buf][col][8 * hi];
      bfr.q[1] = *(const uint4*)&Bt[buf][col][16 + 8 * hi];
      acc[0][bc] = __builtin_amdgcn_wmma_f32_16x16x32_bf16(
          false, a0.v, false, bfr.v, (short)0, acc[0][bc], false, false);
      acc[1][bc] = __builtin_amdgcn_wmma_f32_16x16x32_bf16(
          false, a1.v, false, bfr.v, (short)0, acc[1][bc], false, false);
    }
    if (nxt) { stage(buf ^ 1); buf ^= 1; }
  }

  #pragma unroll
  for (int bc = 0; bc < 4; ++bc) {
    const int col = ncol + cs + bc * 16 + lr;
    const float bv = bias[(long long)e * biasStride + ff0 + col];
    #pragma unroll
    for (int ar = 0; ar < 2; ++ar)
      #pragma unroll
      for (int r = 0; r < 8; ++r) {               // C/D: M = r + 8*hi, N = lr
        const int row = m0 + rs + ar * 16 + r + 8 * hi;
        float v = acc[ar][bc][r] + bv;
        H[(long long)row * ldh + col] = bf1(v > 0.f ? v : 0.f);
      }
  }
}

// ---------------------------------------------------------------------------
// GEMM pass 2: C[rows, D] += H[rows, chunk] * W[ff0:+CHUNK, :]
// C pre-initialized with bias (b2 / d2b) by init kernels.
// ---------------------------------------------------------------------------
__global__ __launch_bounds__(256) void gemm_ffn2(
    const unsigned short* __restrict__ H, int ldh,
    const float* __restrict__ W, long long wStride, int ldw,
    float* __restrict__ C, int ldc,
    const int* __restrict__ offsets, int rowsTotal, int kf0)
{
  __shared__ __align__(16) unsigned short As[2][BM][LDSB];
  __shared__ __align__(16) unsigned short Bt[2][BN][LDSB];

  const int m0 = blockIdx.x * BM;
  int e = 0;
  if (offsets) {
    if (m0 >= offsets[NEXP]) return;
    while (offsets[e + 1] <= m0) ++e;
  } else if (m0 >= rowsTotal) return;

  const int ncol = blockIdx.y * BN;
  const int tid = threadIdx.x, lane = tid & 31, lr = lane & 15, hi = lane >> 4;
  const int wv = tid >> 5;
  const int rs = (wv >> 1) * 32, cs = (wv & 1) * 64;

  const int arow = tid >> 1, acb = (tid & 1) * 16;
  const unsigned short* Ap = H + (long long)(m0 + arow) * ldh + acb;
  const int kq = tid >> 5, nb = (tid & 31) * 4;
  const float* Bp = W + (long long)e * wStride + (long long)kf0 * ldw + ncol + nb;

  uint4 qa0, qa1; float rb[4][4];
  auto loadA = [&](int k0) {                      // H already bf16: 2x b128
    qa0 = *(const uint4*)(Ap + k0);
    qa1 = *(const uint4*)(Ap + k0 + 8);
  };
  auto loadB = [&](int k0) {
    #pragma unroll
    for (int i = 0; i < 4; ++i) {
      float4 v = *(const float4*)(Bp + (long long)(k0 + kq * 4 + i) * ldw);
      rb[i][0] = v.x; rb[i][1] = v.y; rb[i][2] = v.z; rb[i][3] = v.w;
    }
  };
  auto stage = [&](int b) {
    *(uint4*)&As[b][arow][acb]     = qa0;
    *(uint4*)&As[b][arow][acb + 8] = qa1;
    #pragma unroll
    for (int j = 0; j < 4; ++j) {
      uint2 w;
      w.x = pk2(rb[0][j], rb[1][j]);
      w.y = pk2(rb[2][j], rb[3][j]);
      *(uint2*)&Bt[b][nb + j][kq * 4] = w;
    }
  };

  f32x8 acc[2][4];
  #pragma unroll
  for (int ar = 0; ar < 2; ++ar)                  // seed accumulators from C
    #pragma unroll
    for (int bc = 0; bc < 4; ++bc)
      #pragma unroll
      for (int r = 0; r < 8; ++r)
        acc[ar][bc][r] =
            C[(long long)(m0 + rs + ar * 16 + r + 8 * hi) * ldc + ncol + cs + bc * 16 + lr];

  loadA(0); loadB(0); stage(0);
  int buf = 0;
  for (int k0 = 0; k0 < CHUNK; k0 += BK) {
    const bool nxt = (k0 + BK) < CHUNK;
    if (nxt) { loadA(k0 + BK); loadB(k0 + BK); }
    if (k0 + 2 * BK < CHUNK) {
      __builtin_prefetch(Ap + k0 + 2 * BK, 0, 1);
      __builtin_prefetch(Bp + (long long)(k0 + 2 * BK + kq * 4) * ldw, 0, 1);
    }
    __syncthreads();

    FragU a0, a1;
    a0.q[0] = *(const uint4*)&As[buf][rs + lr][8 * hi];
    a0.q[1] = *(const uint4*)&As[buf][rs + lr][16 + 8 * hi];
    a1.q[0] = *(const uint4*)&As[buf][rs + 16 + lr][8 * hi];
    a1.q[1] = *(const uint4*)&As[buf][rs + 16 + lr][16 + 8 * hi];
    #pragma unroll
    for (int bc = 0; bc < 4; ++bc) {
      FragU bfr;
      const int col = cs + bc * 16 + lr;
      bfr.q[0] = *(const uint4*)&Bt[buf][col][8 * hi];
      bfr.q[1] = *(const uint4*)&Bt[buf][col][16 + 8 * hi];
      acc[0][bc] = __builtin_amdgcn_wmma_f32_16x16x32_bf16(
          false, a0.v, false, bfr.v, (short)0, acc[0][bc], false, false);
      acc[1][bc] = __builtin_amdgcn_wmma_f32_16x16x32_bf16(
          false, a1.v, false, bfr.v, (short)0, acc[1][bc], false, false);
    }
    if (nxt) { stage(buf ^ 1); buf ^= 1; }
  }

  #pragma unroll
  for (int ar = 0; ar < 2; ++ar)
    #pragma unroll
    for (int bc = 0; bc < 4; ++bc)
      #pragma unroll
      for (int r = 0; r < 8; ++r)
        C[(long long)(m0 + rs + ar * 16 + r + 8 * hi) * ldc + ncol + cs + bc * 16 + lr] =
            acc[ar][bc][r];
}

__global__ void combine_out(float* __restrict__ out, const float* __restrict__ rp,
                            const int* __restrict__ tok2slot,
                            const float* __restrict__ tok_w,
                            const float* __restrict__ P,
                            const float* __restrict__ Dacc)
{
  size_t idx = (size_t)blockIdx.x * 256 + threadIdx.x;
  if (idx >= (size_t)T_TOK * DIM) return;
  int t = (int)(idx >> 10), d = (int)(idx & (DIM - 1));
  int s0 = tok2slot[t * 2 + 0], s1 = tok2slot[t * 2 + 1];
  float moe = tok_w[t * 2 + 0] * P[(long long)s0 * DIM + d] +
              tok_w[t * 2 + 1] * P[(long long)s1 * DIM + d];
  out[idx] = rp[t * 2 + 0] * moe + rp[t * 2 + 1] * Dacc[idx];
}

// ---------------------------------------------------------------------------
extern "C" void kernel_launch(void* const* d_in, const int* in_sizes, int n_in,
                              void* d_out, int out_size, void* d_ws, size_t ws_size,
                              hipStream_t stream)
{
  const float* x   = (const float*)d_in[0];
  const float* Wr  = (const float*)d_in[1];
  const float* br  = (const float*)d_in[2];
  const float* Wg  = (const float*)d_in[3];
  const float* bg  = (const float*)d_in[4];
  const float* W1  = (const float*)d_in[5];
  const float* b1  = (const float*)d_in[6];
  const float* W2  = (const float*)d_in[7];
  const float* b2  = (const float*)d_in[8];
  const float* D1  = (const float*)d_in[9];
  const float* d1b = (const float*)d_in[10];
  const float* D2  = (const float*)d_in[11];
  const float* d2b = (const float*)d_in[12];
  float* out = (float*)d_out;

  char* ws = (char*)d_ws;
  float* rp        = (float*)(ws + O_RP);
  int*   tok_e     = (int*)  (ws + O_TE);
  float* tok_w     = (float*)(ws + O_TW);
  int*   tok2slot  = (int*)  (ws + O_TS);
  int*   counts    = (int*)  (ws + O_CNT);
  int*   offsets   = (int*)  (ws + O_OFF);
  int*   cursors   = (int*)  (ws + O_CUR);
  int*   tlist     = (int*)  (ws + O_LIST);
  unsigned short* Hbuf = (unsigned short*)(ws + O_H);
  float* pair_out  = (float*)(ws + O_PO);
  float* dense_acc = (float*)(ws + O_DA);

  init_small<<<(SLOTS + 255) / 256, 256, 0, stream>>>(counts, cursors, tlist);
  router_gate<<<T_TOK / 8, 256, 0, stream>>>(x, Wr, br, Wg, bg, rp, tok_e, tok_w, counts);
  make_offsets<<<1, 32, 0, stream>>>(counts, offsets);
  scatter_tokens<<<T_TOK / 256, 256, 0, stream>>>(tok_e, offsets, cursors, tlist, tok2slot);
  init_pairout<<<(unsigned)(((size_t)SLOTS * DIM + 255) / 256), 256, 0, stream>>>(pair_out, b2, offsets);
  init_dense<<<(T_TOK * DIM) / 256, 256, 0, stream>>>(dense_acc, d2b);

  // MoE branch: grouped (expert-segmented) FFN over the 2T routed pairs
  for (int c = 0; c < FFD / CHUNK; ++c) {
    int ff0 = c * CHUNK;
    gemm_ffn1<<<dim3(SLOTS / BM, CHUNK / BN), 256, 0, stream>>>(
        x, DIM, W1, (long long)DIM * FFD, FFD, b1, FFD, Hbuf, CHUNK,
        tlist, offsets, SLOTS, ff0);
    gemm_ffn2<<<dim3(SLOTS / BM, DIM / BN), 256, 0, stream>>>(
        Hbuf, CHUNK, W2, (long long)FFD * DIM, DIM, pair_out, DIM,
        offsets, SLOTS, ff0);
  }
  // Dense branch (same kernels, identity row mapping, single "expert")
  for (int c = 0; c < FFD / CHUNK; ++c) {
    int ff0 = c * CHUNK;
    gemm_ffn1<<<dim3(T_TOK / BM, CHUNK / BN), 256, 0, stream>>>(
        x, DIM, D1, 0, FFD, d1b, 0, Hbuf, CHUNK,
        nullptr, nullptr, T_TOK, ff0);
    gemm_ffn2<<<dim3(T_TOK / BM, DIM / BN), 256, 0, stream>>>(
        Hbuf, CHUNK, D2, 0, DIM, dense_acc, DIM,
        nullptr, T_TOK, ff0);
  }
  combine_out<<<(T_TOK * DIM) / 256, 256, 0, stream>>>(out, rp, tok2slot, tok_w,
                                                       pair_out, dense_acc);
}
